// ModulatedDeformableConv2d_65137474011681
// MI455X (gfx1250) — compile-verified, matrix operand order checked by reading, not run on hardware
//
#include <hip/hip_runtime.h>
#include <hip/hip_bf16.h>

typedef __attribute__((ext_vector_type(16))) _Float16 v16h;
typedef __attribute__((ext_vector_type(8)))  float    v8f;

#define Bn   4
#define Cn   64
#define Hn   128
#define Wn   128
#define On   64
#define K2n  9
#define KKn  576            // C*K2
#define Pn   (Hn*Wn)        // 16384 pixels per batch
#define NT   32             // pixels per block (two 16-wide N tiles)
#define SROW 608            // padded LDS row stride (halves) for sampled tile

// ---------------------------------------------------------------------------
// Kernel 0: repack fp32 weights (O, C, K2) -> f16 flat [o][c*9+k2]
// ---------------------------------------------------------------------------
__global__ void prep_weights(const float* __restrict__ w, _Float16* __restrict__ wh) {
    int i = blockIdx.x * blockDim.x + threadIdx.x;
    if (i < On * KKn) wh[i] = (_Float16)w[i];
}

// ---------------------------------------------------------------------------
// Kernel 1: offset conv (18 ch) + modulation conv (9 ch, sigmoid, mean)
// One 16x16 pixel tile per 256-thread block. All 27*576 weights in LDS.
// ---------------------------------------------------------------------------
__global__ void offset_mod_conv(const float* __restrict__ x,
                                const float* __restrict__ offw,
                                const float* __restrict__ offb,
                                const float* __restrict__ modw,
                                const float* __restrict__ modb,
                                float* __restrict__ dyb,
                                float* __restrict__ dxb,
                                float* __restrict__ modmean) {
    __shared__ float wall[27 * KKn];   // 62208 B
    const int tid = threadIdx.x;

    for (int i = tid; i < 18 * KKn; i += 256) wall[i] = offw[i];
    for (int i = tid; i < 9 * KKn;  i += 256) wall[18 * KKn + i] = modw[i];
    __syncthreads();

    const int b  = blockIdx.x >> 6;
    const int tl = blockIdx.x & 63;
    const int oy = ((tl >> 3) << 4) + (tid >> 4);
    const int ox = ((tl & 7)  << 4) + (tid & 15);

    float acc[27];
#pragma unroll
    for (int j = 0; j < 27; ++j) acc[j] = 0.f;

    const float* xb = x + (size_t)b * Cn * Pn;
    for (int c = 0; c < Cn; ++c) {
        const float* xc = xb + c * Pn;
#pragma unroll
        for (int t = 0; t < 9; ++t) {
            const int iy = oy - 1 + (t / 3);
            const int ix = ox - 1 + (t % 3);
            const bool ok = (iy >= 0) & (iy < Hn) & (ix >= 0) & (ix < Wn);
            const float v = ok ? xc[iy * Wn + ix] : 0.f;
            const int wb = c * 9 + t;
#pragma unroll
            for (int j = 0; j < 27; ++j)
                acc[j] = fmaf(v, wall[j * KKn + wb], acc[j]);
        }
    }

    const int pix = oy * Wn + ox;
    float msum = 0.f;
#pragma unroll
    for (int k2 = 0; k2 < K2n; ++k2) {
        dyb[((b * K2n + k2) * Pn) + pix] = acc[k2]     + offb[k2];
        dxb[((b * K2n + k2) * Pn) + pix] = acc[9 + k2] + offb[9 + k2];
        const float z = acc[18 + k2] + modb[k2];
        msum += 1.f / (1.f + __expf(-z));
    }
    modmean[b * Pn + pix] = msum * (1.f / 9.f);
}

// ---------------------------------------------------------------------------
// Kernel 2: bilinear sampling into LDS (f16) + WMMA contraction + epilogue
// 128 threads = 4 wave32; block handles 32 pixels x all 64 output channels.
// Each wave owns a 16(M) x 32(N) output strip: A fragment reused by 2 WMMAs.
// ---------------------------------------------------------------------------
__global__ void deform_wmma(const float* __restrict__ x,
                            const _Float16* __restrict__ wh,
                            const float* __restrict__ dyb,
                            const float* __restrict__ dxb,
                            const float* __restrict__ modmean,
                            const float* __restrict__ bias,
                            float* __restrict__ out) {
    __shared__ __align__(16) _Float16 S2[NT * SROW];   // sampled tile, N-major
    __shared__ int   sY0[K2n * NT];
    __shared__ int   sX0[K2n * NT];
    __shared__ float sWY[K2n * NT];
    __shared__ float sWX[K2n * NT];

    const int tid   = threadIdx.x;
    const int b     = blockIdx.x >> 9;           // 512 pixel tiles per batch
    const int pbase = (blockIdx.x & 511) << 5;   // 32 pixels per block

    // --- phase 1: sampling coordinates for 9*32 (k2, pixel) pairs ----------
    for (int idx = tid; idx < K2n * NT; idx += 128) {
        const int k2  = idx >> 5;
        const int p   = idx & 31;
        const int pix = pbase + p;
        const float oy = (float)(pix >> 7);
        const float ox = (float)(pix & 127);
        const float sy = oy + dyb[((b * K2n + k2) * Pn) + pix];
        const float sx = ox + dxb[((b * K2n + k2) * Pn) + pix];
        const float fy = floorf(sy);
        const float fx = floorf(sx);
        sY0[idx] = (int)fy;  sX0[idx] = (int)fx;
        sWY[idx] = sy - fy;  sWX[idx] = sx - fx;
    }
    __syncthreads();

    // --- phase 2: build 576x32 sampled tile (f16) in LDS -------------------
    for (int idx = tid; idx < KKn * NT; idx += 128) {
        const int p  = idx & 31;
        const int kc = idx >> 5;           // kc = c*9 + k2
        const int c  = kc / 9;
        const int k2 = kc - c * 9;
        const int pi = (k2 << 5) + p;

        const int   y0 = sY0[pi], x0 = sX0[pi];
        const float wy = sWY[pi], wx = sWX[pi];
        const float* xc = x + ((size_t)(b * Cn + c) * Pn);

        const bool vy0 = (y0 >= 0) & (y0 < Hn), vy1 = (y0 + 1 >= 0) & (y0 + 1 < Hn);
        const bool vx0 = (x0 >= 0) & (x0 < Wn), vx1 = (x0 + 1 >= 0) & (x0 + 1 < Wn);
        const int y0c = min(max(y0, 0), Hn - 1),     y1c = min(max(y0 + 1, 0), Hn - 1);
        const int x0c = min(max(x0, 0), Wn - 1),     x1c = min(max(x0 + 1, 0), Wn - 1);

        const float g00 = (vy0 & vx0) ? xc[y0c * Wn + x0c] : 0.f;
        const float g01 = (vy0 & vx1) ? xc[y0c * Wn + x1c] : 0.f;
        const float g10 = (vy1 & vx0) ? xc[y1c * Wn + x0c] : 0.f;
        const float g11 = (vy1 & vx1) ? xc[y1c * Wn + x1c] : 0.f;

        const float val = (1.f - wy) * ((1.f - wx) * g00 + wx * g01)
                        +        wy  * ((1.f - wx) * g10 + wx * g11);
        S2[p * SROW + kc] = (_Float16)val;
    }
    __syncthreads();

    // --- phase 3: WMMA  D[16x32] = A(weights 16x576) x B(sampled 576x32) ---
    const int wave = tid >> 5;
    const int lane = tid & 31;
    const int g    = lane >> 4;    // lane half
    const int m    = lane & 15;    // A row / B col / D col

    v8f acc0 = {};
    v8f acc1 = {};
    const unsigned* wrow  = (const unsigned*)(wh + (wave * 16 + m) * KKn);
    const unsigned* srow0 = (const unsigned*)(&S2[m * SROW]);          // N tile 0
    const unsigned* srow1 = (const unsigned*)(&S2[(16 + m) * SROW]);   // N tile 1

    for (int kk = 0; kk < 18; ++kk) {
        const int base = kk * 32;
        union { v16h h; unsigned u[8]; } A, B0, B1;
        // 16-bit A layout: VGPR v<4 -> K = base + 8g + 2v ; v>=4 -> +16
#pragma unroll
        for (int v = 0; v < 4; ++v) {
            const int k0 = (base + 8 * g + 2 * v) >> 1;
            const int k1 = (base + 16 + 8 * g + 2 * v) >> 1;
            A.u[v]      = wrow[k0];   A.u[4 + v]  = wrow[k1];
            B0.u[v]     = srow0[k0];  B0.u[4 + v] = srow0[k1];
            B1.u[v]     = srow1[k0];  B1.u[4 + v] = srow1[k1];
        }
        acc0 = __builtin_amdgcn_wmma_f32_16x16x32_f16(
                   false, A.h, false, B0.h, (short)0, acc0, false, false);
        acc1 = __builtin_amdgcn_wmma_f32_16x16x32_f16(
                   false, A.h, false, B1.h, (short)0, acc1, false, false);
    }

    // --- epilogue: out = acc * modmean + bias  (C/D layout: VGPR r -> M=r+8g)
    const int pix0 = pbase + m;
    const int pix1 = pbase + 16 + m;
    const float mm0 = modmean[b * Pn + pix0];
    const float mm1 = modmean[b * Pn + pix1];
#pragma unroll
    for (int r = 0; r < 8; ++r) {
        const int o = wave * 16 + g * 8 + r;
        const size_t rowoff = ((size_t)(b * On + o) << 14);
        out[rowoff + pix0] = acc0[r] * mm0 + bias[o];
        out[rowoff + pix1] = acc1[r] * mm1 + bias[o];
    }
}

// ---------------------------------------------------------------------------
extern "C" void kernel_launch(void* const* d_in, const int* in_sizes, int n_in,
                              void* d_out, int out_size, void* d_ws, size_t ws_size,
                              hipStream_t stream) {
    const float* x    = (const float*)d_in[0];
    const float* w    = (const float*)d_in[1];
    const float* bias = (const float*)d_in[2];
    const float* offw = (const float*)d_in[3];
    const float* offb = (const float*)d_in[4];
    const float* modw = (const float*)d_in[5];
    const float* modb = (const float*)d_in[6];

    char* ws = (char*)d_ws;
    _Float16* wh      = (_Float16*)ws;                             //   73728 B
    float*    dyb     = (float*)(ws + 73728);                      // 2359296 B
    float*    dxb     = (float*)(ws + 73728 + 2359296);            // 2359296 B
    float*    mmean   = (float*)(ws + 73728 + 2 * 2359296);        //  262144 B

    prep_weights   <<<(On * KKn + 255) / 256, 256, 0, stream>>>(w, wh);
    offset_mod_conv<<<Bn * 64,  256, 0, stream>>>(x, offw, offb, modw, modb,
                                                  dyb, dxb, mmean);
    deform_wmma    <<<Bn * 512, 128, 0, stream>>>(x, wh, dyb, dxb, mmean,
                                                  bias, (float*)d_out);
}
